// ResetLSTM_67362267070747
// MI455X (gfx1250) — matrix operand order repaired and checked by reading, not compiled, not run on hardware
//
#include <hip/hip_runtime.h>

#define T_STEPS 512
#define B_DIM 256
#define D_DIM 256
#define H_DIM 256
#define NWG 32

typedef __attribute__((ext_vector_type(16))) __bf16 v16bf;
typedef __attribute__((ext_vector_type(8)))  float  v8f;
typedef __attribute__((ext_vector_type(8)))  unsigned short u16x8;

__device__ __forceinline__ unsigned short f2bf(float f) {
  union { float f; unsigned u; } cv; cv.f = f;
  unsigned u = cv.u;
  u += 0x7fffu + ((u >> 16) & 1u);   // round-to-nearest-even
  return (unsigned short)(u >> 16);
}

__device__ __forceinline__ float sigf(float x) {
  return 1.0f / (1.0f + __expf(-x));
}

// ---------------------------------------------------------------------------
// Init: transpose+convert Wi/Wh to K-major bf16, zero c and h0, reset barrier.
// ---------------------------------------------------------------------------
__global__ void lstm_init(const float* __restrict__ Wi, const float* __restrict__ Wh,
                          unsigned short* __restrict__ WiT, unsigned short* __restrict__ WhT,
                          float* __restrict__ c0, unsigned short* __restrict__ h0,
                          unsigned int* __restrict__ syncc) {
  int idx = blockIdx.x * blockDim.x + threadIdx.x;
  if (idx == 0) syncc[0] = 0u;
  if (idx < 262144) {                       // WiT[n][k] = Wi[k][n]
    int n = idx >> 8, k = idx & 255;
    WiT[idx] = f2bf(Wi[k * (4 * H_DIM) + n]);
  } else if (idx < 524288) {                // WhT[n][k] = Wh[k][n]
    int j = idx - 262144;
    int n = j >> 8, k = j & 255;
    WhT[j] = f2bf(Wh[k * (4 * H_DIM) + n]);
  } else if (idx < 589824) {
    c0[idx - 524288] = 0.0f;
  } else if (idx < 655360) {
    h0[idx - 589824] = 0;
  }
}

// ---------------------------------------------------------------------------
// Pre-convert all of x to bf16 (fully parallel, removes conversion VALU from
// the serial scan loop and halves x HBM traffic). 8 elements per thread.
// ---------------------------------------------------------------------------
__global__ void lstm_xconv(const float* __restrict__ x,
                           unsigned short* __restrict__ xbf) {
  size_t i = ((size_t)blockIdx.x * blockDim.x + threadIdx.x) * 8;
  float4 a = *(const float4*)(x + i);
  float4 b = *(const float4*)(x + i + 4);
  union { unsigned short u[8]; u16x8 v; } o;
  o.u[0] = f2bf(a.x); o.u[1] = f2bf(a.y); o.u[2] = f2bf(a.z); o.u[3] = f2bf(a.w);
  o.u[4] = f2bf(b.x); o.u[5] = f2bf(b.y); o.u[6] = f2bf(b.z); o.u[7] = f2bf(b.w);
  *(u16x8*)(xbf + i) = o.v;
}

// ---------------------------------------------------------------------------
// Persistent scan kernel: 32 WGs x 256 threads (8 waves). One grid barrier
// per time step. Weights resident in LDS (64KB/WG) for all 512 steps.
// XBF16: x pre-converted to bf16 in workspace (fast path); otherwise convert
// x fragments on the fly.
// ---------------------------------------------------------------------------
template <bool XBF16>
__global__ void __launch_bounds__(256, 1) lstm_scan(
    const float* __restrict__ x, const unsigned short* __restrict__ xbf,
    const unsigned char* __restrict__ resets,
    const float* __restrict__ bias,
    const unsigned short* __restrict__ WiT, const unsigned short* __restrict__ WhT,
    float* __restrict__ c, unsigned short* __restrict__ h0,
    unsigned short* __restrict__ h1,
    float* __restrict__ out, unsigned int* __restrict__ syncc)
{
  // [Wi/Wh][gate][col-in-slice][K]  = 2*4*16*256 bf16 = 64KB
  __shared__ __align__(32) unsigned short ldsW[2][4][16][256];

  const int tid    = threadIdx.x;
  const int wg     = blockIdx.x;
  const int colsl  = wg & 15;      // 16 column slices of H
  const int rowgp  = wg >> 4;      // 2 row groups
  const int wave   = tid >> 5;
  const int lane   = tid & 31;
  const int laneN  = lane & 15;
  const int laneHi = lane >> 4;
  const int m0 = (rowgp * 8 + wave) * 16;   // row tile of B
  const int n0 = colsl * 16;                // col tile within H

  // Stage this WG's weight slices into LDS (once, reused for 512 steps).
  for (int i = tid; i < 2 * 4 * 16 * 256; i += 256) {
    int sel = i >> 14;
    int rem = i & 16383;
    int g  = rem >> 12;
    int cn = (rem >> 8) & 15;
    int k  = rem & 255;
    const unsigned short* src = sel ? WhT : WiT;
    ldsW[sel][g][cn][k] = src[(g * H_DIM + n0 + cn) * 256 + k];
  }
  __syncthreads();

  float bgate[4];
#pragma unroll
  for (int g = 0; g < 4; ++g) bgate[g] = bias[g * H_DIM + n0 + laneN];

  float* cf = out;
  float* hf = out + B_DIM * H_DIM;
  float* ys = out + 2 * (B_DIM * H_DIM);
  unsigned short* hbuf[2] = { h0, h1 };

  for (int t = 0; t < T_STEPS; ++t) {
    const unsigned short* hcur = hbuf[t & 1];
    unsigned short*       hnxt = hbuf[(t + 1) & 1];
    const float*          xt   = x   + (size_t)t * B_DIM * D_DIM;
    const unsigned short* xbt  = xbf + (size_t)t * B_DIM * D_DIM;
    const unsigned char*  rt   = resets + (size_t)t * B_DIM;

    const int m    = m0 + laneN;     // A-fragment row owned by this lane
    const int hi8  = laneHi * 8;
    const int hi16 = laneHi * 16;
    const unsigned char rstA = rt[m];

    if (t + 1 < T_STEPS) {           // warm L2 for next step's x rows
      if (XBF16)
        __builtin_prefetch(xbt + B_DIM * D_DIM + (size_t)m * D_DIM, 0, 0);
      else
        __builtin_prefetch(xt + B_DIM * D_DIM + (size_t)m * D_DIM, 0, 0);
    }

    v8f acc[4];
#pragma unroll
    for (int g = 0; g < 4; ++g)
#pragma unroll
      for (int r = 0; r < 8; ++r) acc[g][r] = bgate[g];

    union { unsigned short u[16]; v16bf v; u16x8 h8[2]; } A;

#pragma unroll
    for (int kc = 0; kc < 16; ++kc) {
      if (kc < 8) {
        // A from x[t]. 16-bit A layout: lane<16 K=j|j+16, lane>=16 shifted
        // by 8 -> two runs of 8 contiguous elements.
        const int kx = kc * 32;
        if (XBF16) {
          const unsigned short* xs = xbt + (size_t)m * D_DIM + kx + hi8;
          A.h8[0] = *(const u16x8*)(xs);
          A.h8[1] = *(const u16x8*)(xs + 16);
        } else {
          const float* xs = xt + (size_t)m * D_DIM + kx + hi8;
          float4 f0 = *(const float4*)(xs);
          float4 f1 = *(const float4*)(xs + 4);
          float4 f2 = *(const float4*)(xs + 16);
          float4 f3 = *(const float4*)(xs + 20);
          A.u[0]=f2bf(f0.x);  A.u[1]=f2bf(f0.y);  A.u[2]=f2bf(f0.z);  A.u[3]=f2bf(f0.w);
          A.u[4]=f2bf(f1.x);  A.u[5]=f2bf(f1.y);  A.u[6]=f2bf(f1.z);  A.u[7]=f2bf(f1.w);
          A.u[8]=f2bf(f2.x);  A.u[9]=f2bf(f2.y);  A.u[10]=f2bf(f2.z); A.u[11]=f2bf(f2.w);
          A.u[12]=f2bf(f3.x); A.u[13]=f2bf(f3.y); A.u[14]=f2bf(f3.z); A.u[15]=f2bf(f3.w);
        }
      } else {
        // A from h (already bf16); apply reset mask before the GEMM.
        const int kh = (kc - 8) * 32;
        const unsigned short* hs = hcur + (size_t)m * H_DIM + kh + hi8;
        if (rstA) {
#pragma unroll
          for (int j = 0; j < 16; ++j) A.u[j] = 0;
        } else {
          A.h8[0] = *(const u16x8*)(hs);
          A.h8[1] = *(const u16x8*)(hs + 16);
        }
      }
      const int sel  = kc >> 3;
      const int koff = (kc & 7) * 32 + hi8;  // B layout: K = j|j+16, lane>=16 shifted by 8
#pragma unroll
      for (int g = 0; g < 4; ++g) {
        union { u16x8 h8[2]; v16bf v; } Bu;
        Bu.h8[0] = *(const u16x8*)&ldsW[sel][g][laneN][koff];
        Bu.h8[1] = *(const u16x8*)&ldsW[sel][g][laneN][koff + 16];
        const v16bf bmat = Bu.v;
        acc[g] = __builtin_amdgcn_wmma_f32_16x16x32_bf16(
            false, A.v, false, bmat, (short)0, acc[g], false, false);
      }
    }

    // Pointwise cell update. C/D layout: lane owns column n0+laneN,
    // rows m0 + laneHi*8 + r for r in 0..7.
    const int n = n0 + laneN;
#pragma unroll
    for (int r = 0; r < 8; ++r) {
      const int mr  = m0 + laneHi * 8 + r;
      const int idx = mr * H_DIM + n;
      float iv = sigf(acc[0][r]);
      float fv = sigf(acc[1][r]);
      float gv = tanhf(acc[2][r]);
      float ov = sigf(acc[3][r]);
      float cold = rt[mr] ? 0.0f : c[idx];
      float cn = fv * cold + iv * gv;
      float hn = ov * tanhf(cn);
      c[idx] = cn;
      hnxt[idx] = f2bf(hn);
      ys[(size_t)t * (B_DIM * H_DIM) + idx] = hn;
      if (t == T_STEPS - 1) { cf[idx] = cn; hf[idx] = hn; }
    }

    // Grid barrier (monotonic counter, release/acquire through L2).
    __threadfence();
    __syncthreads();
    if (tid == 0) {
      __hip_atomic_fetch_add(syncc, 1u, __ATOMIC_RELEASE, __HIP_MEMORY_SCOPE_AGENT);
      const unsigned target = (unsigned)NWG * (unsigned)(t + 1);
      while (__hip_atomic_load(syncc, __ATOMIC_ACQUIRE, __HIP_MEMORY_SCOPE_AGENT) < target) {
        __builtin_amdgcn_s_sleep(1);
      }
    }
    __syncthreads();
  }
}

// ---------------------------------------------------------------------------
extern "C" void kernel_launch(void* const* d_in, const int* in_sizes, int n_in,
                              void* d_out, int out_size, void* d_ws, size_t ws_size,
                              hipStream_t stream) {
  const float*         x      = (const float*)d_in[0];
  const unsigned char* resets = (const unsigned char*)d_in[1];
  const float*         Wi     = (const float*)d_in[2];
  const float*         Wh     = (const float*)d_in[3];
  const float*         b      = (const float*)d_in[4];

  char* ws = (char*)d_ws;
  unsigned short* WiT  = (unsigned short*)(ws);              // 1024*256 bf16
  unsigned short* WhT  = (unsigned short*)(ws + 524288);     // 1024*256 bf16
  float*          c    = (float*)(ws + 1048576);             // 256*256 f32
  unsigned short* h0   = (unsigned short*)(ws + 1310720);    // 256*256 bf16
  unsigned short* h1   = (unsigned short*)(ws + 1441792);    // 256*256 bf16
  unsigned int*   sy   = (unsigned int*)(ws + 1572864);
  unsigned short* xbf  = (unsigned short*)(ws + 1703936);    // T*B*D bf16 (67MB)

  const size_t need_xbf = 1703936u + (size_t)T_STEPS * B_DIM * D_DIM * 2u;

  lstm_init<<<2560, 256, 0, stream>>>(Wi, Wh, WiT, WhT, c, h0, sy);

  if (ws_size >= need_xbf) {
    // Fast path: pre-convert x once (parallel), scan loop does pure bf16 loads.
    const size_t nx = (size_t)T_STEPS * B_DIM * D_DIM;       // 33,554,432
    lstm_xconv<<<(unsigned)(nx / 8 / 256), 256, 0, stream>>>(x, xbf);
    lstm_scan<true><<<NWG, 256, 0, stream>>>(x, xbf, resets, b, WiT, WhT,
                                             c, h0, h1, (float*)d_out, sy);
  } else {
    lstm_scan<false><<<NWG, 256, 0, stream>>>(x, xbf, resets, b, WiT, WhT,
                                              c, h0, h1, (float*)d_out, sy);
  }
}
